// LagAwarePeerBlock_26783416058525
// MI455X (gfx1250) — matrix operand order repaired and checked by reading, not compiled, not run on hardware
//
#include <hip/hip_runtime.h>
#include <hip/hip_bf16.h>
#include <math.h>

typedef __attribute__((ext_vector_type(16))) __bf16 v16bf;
typedef __attribute__((ext_vector_type(8)))  __bf16 v8bf;
typedef __attribute__((ext_vector_type(8)))  float  v8f;

namespace lap {

constexpr int B = 4, N = 64, T = 256, H = 256, L = 5;
constexpr int NL   = N * L;        // 320
constexpr int TOPK = 8;
constexpr int BT   = B * T;        // 1024
constexpr int MPEER = B * N * T;   // 65536
constexpr float LN_EPS = 1e-5f;

constexpr int WPAD   = 8;              // bf16 pad per LDS row (16B -> 4-bank skew)
constexpr int WROW   = H + WPAD;       // 264 bf16 = 528 B per LDS row

__constant__ int LAGS_D[L] = {1, 5, 10, 21, 30};

// ---------------------------------------------------------------------------
// fp32 -> bf16 pre-conversion (8 elements / thread, b128 in, b128 out)
// ---------------------------------------------------------------------------
__global__ void __launch_bounds__(256) f32_to_bf16_kernel(
    const float* __restrict__ src, __bf16* __restrict__ dst, int n8)
{
  const int i = blockIdx.x * 256 + threadIdx.x;
  if (i >= n8) return;
  const float4 x0 = *reinterpret_cast<const float4*>(src + (size_t)i * 8);
  const float4 x1 = *reinterpret_cast<const float4*>(src + (size_t)i * 8 + 4);
  v8bf o;
  o[0]=(__bf16)x0.x; o[1]=(__bf16)x0.y; o[2]=(__bf16)x0.z; o[3]=(__bf16)x0.w;
  o[4]=(__bf16)x1.x; o[5]=(__bf16)x1.y; o[6]=(__bf16)x1.z; o[7]=(__bf16)x1.w;
  *reinterpret_cast<v8bf*>(dst + (size_t)i * 8) = o;
}

// ---------------------------------------------------------------------------
// Fragment loaders.
// A (16x32 bf16): lane L holds row M=L&15, K = lhalf*8 + {0..7} and +16.
// ---------------------------------------------------------------------------
__device__ __forceinline__ v16bf load_afrag(const __bf16* arow, int kk, int lhalf)
{
  const __bf16* ap = arow + kk + lhalf * 8;
  const v8bf x0 = *reinterpret_cast<const v8bf*>(ap);
  const v8bf x1 = *reinterpret_cast<const v8bf*>(ap + 16);
  return __builtin_shufflevector(x0, x1, 0,1,2,3,4,5,6,7,8,9,10,11,12,13,14,15);
}

__device__ __forceinline__ v16bf load_afrag(const float* arow, int kk, int lhalf)
{
  const float* ap = arow + kk + lhalf * 8;
  const float4 x0 = *reinterpret_cast<const float4*>(ap + 0);
  const float4 x1 = *reinterpret_cast<const float4*>(ap + 4);
  const float4 x2 = *reinterpret_cast<const float4*>(ap + 16);
  const float4 x3 = *reinterpret_cast<const float4*>(ap + 20);
  v16bf a;
  a[0]=(__bf16)x0.x;  a[1]=(__bf16)x0.y;  a[2]=(__bf16)x0.z;  a[3]=(__bf16)x0.w;
  a[4]=(__bf16)x1.x;  a[5]=(__bf16)x1.y;  a[6]=(__bf16)x1.z;  a[7]=(__bf16)x1.w;
  a[8]=(__bf16)x2.x;  a[9]=(__bf16)x2.y;  a[10]=(__bf16)x2.z; a[11]=(__bf16)x2.w;
  a[12]=(__bf16)x3.x; a[13]=(__bf16)x3.y; a[14]=(__bf16)x3.z; a[15]=(__bf16)x3.w;
  return a;
}

// ---------------------------------------------------------------------------
// C[M,256] = act( A[M,256] @ W^T + bias ), bf16 WMMA fp32 accumulate.
// Block = 8 waves sharing ONE 64-col weight strip staged in LDS via
// global_load_async_to_lds_b128 (ASYNCcnt).  Each wave: 32(M) x 64(N) tile.
// ACT: 0 = identity, 1 = ELU.  AT: float | __bf16.  OutT: float | __bf16.
// Requires M % 256 == 0.
// ---------------------------------------------------------------------------
template <int ACT, typename AT, typename OutT>
__global__ void __launch_bounds__(256) gemm_xwt_kernel(
    const AT* __restrict__ A, const __bf16* __restrict__ Wb,
    const float* __restrict__ bias, OutT* __restrict__ C, int M)
{
  __shared__ __bf16 sW[64 * WROW];          // 64 weight rows, padded (33 KB)

  const int tid    = threadIdx.x;
  const int lane   = tid & 31;
  const int wave   = tid >> 5;              // 0..7
  const int strip  = blockIdx.x & 3;        // which 64-col strip of W
  const int mgroup = blockIdx.x >> 2;
  const int mtile  = mgroup * 8 + wave;     // 32 output rows per wave
  (void)M;

  // ---- async fill of the weight strip into LDS (gfx1250 async path) ----
  {
    const char* wsrc = (const char*)(Wb + (size_t)strip * 64 * H);
    const unsigned lds0 = (unsigned)(uintptr_t)(&sW[0]);
    // 64 rows x 512B, 16B chunks: 2048 chunks, 8 per thread
#pragma unroll
    for (int it = 0; it < 8; ++it) {
      const unsigned c     = (unsigned)(it * 256 + tid);
      const unsigned row   = c >> 5;            // 32 chunks per 512B row
      const unsigned inrow = (c & 31) * 16;
      const unsigned lds_addr = lds0 + row * (WROW * 2) + inrow;
      const unsigned long long ga =
          (unsigned long long)(uintptr_t)(wsrc + row * 512 + inrow);
      asm volatile("global_load_async_to_lds_b128 %0, %1, off"
                   :: "v"(lds_addr), "v"(ga) : "memory");
    }
    asm volatile("s_wait_asynccnt 0" ::: "memory");
    __syncthreads();
  }

  const int lrow  = lane & 15;
  const int lhalf = lane >> 4;              // 0 or 1
  const AT* arow0 = A + (size_t)(mtile * 32 + lrow) * H;
  const AT* arow1 = arow0 + (size_t)16 * H;

  v8f acc[2][4];
#pragma unroll
  for (int j = 0; j < 4; ++j) {
    const float bv = bias[strip * 64 + j * 16 + lrow];
#pragma unroll
    for (int e = 0; e < 8; ++e) { acc[0][j][e] = bv; acc[1][j][e] = bv; }
  }

  for (int kk = 0; kk < H; kk += 32) {
    const v16bf a0 = load_afrag(arow0, kk, lhalf);
    const v16bf a1 = load_afrag(arow1, kk, lhalf);
#pragma unroll
    for (int j = 0; j < 4; ++j) {
      // B fragment from LDS: lane L col n=j*16+lrow, K = kk + lhalf*8 + {0..7} and +16
      const __bf16* bp = &sW[(j * 16 + lrow) * WROW + kk + lhalf * 8];
      const v8bf y0 = *reinterpret_cast<const v8bf*>(bp);
      const v8bf y1 = *reinterpret_cast<const v8bf*>(bp + 16);
      const v16bf bm =
          __builtin_shufflevector(y0, y1, 0,1,2,3,4,5,6,7,8,9,10,11,12,13,14,15);
      acc[0][j] = __builtin_amdgcn_wmma_f32_16x16x32_bf16(
          false, a0, false, bm, (short)0, acc[0][j], false, false);
      acc[1][j] = __builtin_amdgcn_wmma_f32_16x16x32_bf16(
          false, a1, false, bm, (short)0, acc[1][j], false, false);
    }
  }

  // epilogue: lane L col n=L&15, element e row = e + lhalf*8 (+16 for half 1)
#pragma unroll
  for (int half = 0; half < 2; ++half) {
    const int m0 = mtile * 32 + half * 16 + lhalf * 8;
#pragma unroll
    for (int j = 0; j < 4; ++j) {
      const int n = strip * 64 + j * 16 + lrow;
#pragma unroll
      for (int e = 0; e < 8; ++e) {
        float v = acc[half][j][e];
        if (ACT == 1) v = (v > 0.f) ? v : (expf(v) - 1.f);
        C[(size_t)(m0 + e) * H + n] = (OutT)v;
      }
    }
  }
}

// ---------------------------------------------------------------------------
// logits[b,t,n,l] = (Q[b,t] . Kp[b,n,t-lag_l]) / 16, masked to -inf.
// One wave per (b,t,n); lane holds 8 contiguous channels.
// ---------------------------------------------------------------------------
__global__ void __launch_bounds__(256) lag_logits_kernel(
    const float* __restrict__ Q,                 // [BT, H]
    const __bf16* __restrict__ Kp,               // [B,N,T,H]
    const unsigned char* __restrict__ peer_mask, // [B,N] (bool)
    float* __restrict__ logits)                  // [BT, NL]
{
  const int lane = threadIdx.x & 31;
  const int gw   = (blockIdx.x * blockDim.x + threadIdx.x) >> 5;
  if (gw >= B * T * N) return;
  const int n  = gw % N;
  const int bt = gw / N;
  const int b  = bt / T;
  const int t  = bt % T;

  float q[8];
  {
    const float4* qp = reinterpret_cast<const float4*>(Q + (size_t)bt * H + lane * 8);
    const float4 q0 = qp[0], q1 = qp[1];
    q[0]=q0.x; q[1]=q0.y; q[2]=q0.z; q[3]=q0.w;
    q[4]=q1.x; q[5]=q1.y; q[6]=q1.z; q[7]=q1.w;
  }
  const bool pm = peer_mask[b * N + n] != 0;

#pragma unroll
  for (int l = 0; l < L; ++l) {
    const int ts = t - LAGS_D[l];
    float outv;
    if (ts >= 0 && pm) {
      const v8bf kv = *reinterpret_cast<const v8bf*>(
          Kp + ((size_t)(b * N + n) * T + ts) * H + lane * 8);
      float d = 0.f;
#pragma unroll
      for (int e = 0; e < 8; ++e) d += q[e] * (float)kv[e];
#pragma unroll
      for (int off = 16; off >= 1; off >>= 1) d += __shfl_xor(d, off);
      outv = d * 0.0625f;   // 1/sqrt(256)
    } else {
      outv = -INFINITY;
    }
    if (lane == 0) logits[(size_t)bt * NL + n * L + l] = outv;
  }
}

// ---------------------------------------------------------------------------
// Per (b,t): top-8 over 320 logits, softmax (with -inf -> -1e9, all-inf -> 0),
// weighted sum of selected V rows. One 256-thread block per (b,t).
// ---------------------------------------------------------------------------
__global__ void __launch_bounds__(256) topk_attend_kernel(
    const float* __restrict__ logits,   // [BT, NL]
    const __bf16* __restrict__ Vp,      // [B,N,T,H]
    float* __restrict__ cs_y)           // [BT, H]
{
  __shared__ float sl[NL];
  __shared__ float rv[256];
  __shared__ int   ri[256];
  __shared__ float s_w[TOPK];
  __shared__ int   s_i[TOPK];

  const int bt  = blockIdx.x;
  const int b   = bt / T, t = bt % T;
  const int tid = threadIdx.x;

  for (int i = tid; i < NL; i += 256) sl[i] = logits[(size_t)bt * NL + i];
  __syncthreads();

  for (int k = 0; k < TOPK; ++k) {
    float best = -INFINITY; int bi = 0;
    for (int i = tid; i < NL; i += 256)
      if (sl[i] > best) { best = sl[i]; bi = i; }
    rv[tid] = best; ri[tid] = bi;
    __syncthreads();
    for (int s = 128; s > 0; s >>= 1) {
      if (tid < s) {
        const float ov = rv[tid + s]; const int oi = ri[tid + s];
        if (ov > rv[tid] || (ov == rv[tid] && oi < ri[tid])) { rv[tid] = ov; ri[tid] = oi; }
      }
      __syncthreads();
    }
    if (tid == 0) { s_w[k] = rv[0]; s_i[k] = ri[0]; sl[ri[0]] = -INFINITY; }
    __syncthreads();
  }

  if (tid == 0) {
    float vals[TOPK]; float mx = -INFINITY; bool allinf = true;
    for (int k = 0; k < TOPK; ++k) {
      float v = s_w[k];
      if (v == -INFINITY) v = -1e9f; else allinf = false;
      vals[k] = v; mx = fmaxf(mx, v);
    }
    float ssum = 0.f;
    for (int k = 0; k < TOPK; ++k) { vals[k] = expf(vals[k] - mx); ssum += vals[k]; }
    const float inv = allinf ? 0.f : (1.f / ssum);
    for (int k = 0; k < TOPK; ++k) s_w[k] = vals[k] * inv;
  }
  __syncthreads();

  float accum = 0.f;
#pragma unroll
  for (int k = 0; k < TOPK; ++k) {
    const int idx = s_i[k];
    const int n = idx / L, l = idx % L;
    int ts = t - LAGS_D[l]; if (ts < 0) ts = 0;
    accum += s_w[k] * (float)Vp[((size_t)(b * N + n) * T + ts) * H + tid];
  }
  cs_y[(size_t)bt * H + tid] = accum;
}

// ---------------------------------------------------------------------------
// y = cs_y + f2 ; LayerNorm(y) * gamma + beta.  One block per (b,t).
// ---------------------------------------------------------------------------
__global__ void __launch_bounds__(256) resid_ln_kernel(
    const float* __restrict__ cs_y, const float* __restrict__ f2,
    const float* __restrict__ gamma, const float* __restrict__ beta,
    float* __restrict__ out)
{
  __shared__ float red[256];
  __shared__ float s_mu, s_var;
  const int bt = blockIdx.x, tid = threadIdx.x;
  const float y = cs_y[(size_t)bt * H + tid] + f2[(size_t)bt * H + tid];

  red[tid] = y; __syncthreads();
  for (int s = 128; s > 0; s >>= 1) { if (tid < s) red[tid] += red[tid + s]; __syncthreads(); }
  if (tid == 0) s_mu = red[0] * (1.f / H);
  __syncthreads();

  const float d = y - s_mu;
  red[tid] = d * d; __syncthreads();
  for (int s = 128; s > 0; s >>= 1) { if (tid < s) red[tid] += red[tid + s]; __syncthreads(); }
  if (tid == 0) s_var = red[0] * (1.f / H);
  __syncthreads();

  out[(size_t)bt * H + tid] = d * rsqrtf(s_var + LN_EPS) * gamma[tid] + beta[tid];
}

} // namespace lap

extern "C" void kernel_launch(void* const* d_in, const int* in_sizes, int n_in,
                              void* d_out, int out_size, void* d_ws, size_t ws_size,
                              hipStream_t stream) {
  using namespace lap;
  (void)in_sizes; (void)n_in; (void)out_size; (void)ws_size;

  const float* target_h = (const float*)d_in[0];
  const float* peer_h   = (const float*)d_in[1];
  const unsigned char* peer_mask = (const unsigned char*)d_in[2];
  const float* wq = (const float*)d_in[3];
  const float* bq = (const float*)d_in[4];
  const float* wk = (const float*)d_in[5];
  const float* bk = (const float*)d_in[6];
  const float* wv = (const float*)d_in[7];
  const float* bv = (const float*)d_in[8];
  const float* w1 = (const float*)d_in[9];
  const float* b1 = (const float*)d_in[10];
  const float* w2 = (const float*)d_in[11];
  const float* b2 = (const float*)d_in[12];
  const float* gamma = (const float*)d_in[13];
  const float* beta  = (const float*)d_in[14];
  float* out = (float*)d_out;

  char* ws = (char*)d_ws;
  size_t off = 0;
  auto alloc = [&](size_t bytes) -> void* {
    void* p = ws + off;
    off += (bytes + 255) & ~(size_t)255;
    return p;
  };
  float*  Q    = (float*) alloc((size_t)BT * H * sizeof(float));       // 1 MB
  __bf16* Kp   = (__bf16*)alloc((size_t)MPEER * H * sizeof(__bf16));   // 32 MB
  __bf16* Vp   = (__bf16*)alloc((size_t)MPEER * H * sizeof(__bf16));   // 32 MB
  __bf16* Pb   = (__bf16*)alloc((size_t)MPEER * H * sizeof(__bf16));   // 32 MB (peer_h bf16)
  float*  lg   = (float*) alloc((size_t)BT * NL * sizeof(float));      // 1.25 MB
  float*  csy  = (float*) alloc((size_t)BT * H * sizeof(float));       // 1 MB
  float*  h1   = (float*) alloc((size_t)BT * H * sizeof(float));       // 1 MB
  float*  f2   = (float*) alloc((size_t)BT * H * sizeof(float));       // 1 MB
  __bf16* wqb  = (__bf16*)alloc((size_t)H * H * sizeof(__bf16));       // 128 KB x5
  __bf16* wkb  = (__bf16*)alloc((size_t)H * H * sizeof(__bf16));
  __bf16* wvb  = (__bf16*)alloc((size_t)H * H * sizeof(__bf16));
  __bf16* w1b  = (__bf16*)alloc((size_t)H * H * sizeof(__bf16));
  __bf16* w2b  = (__bf16*)alloc((size_t)H * H * sizeof(__bf16));

  // --- one-time bf16 conversions (peer activations + all weights) ---
  const int n8peer = MPEER * H / 8;        // 2,097,152
  const int n8w    = H * H / 8;            // 8,192
  f32_to_bf16_kernel<<<n8peer / 256, 256, 0, stream>>>(peer_h, Pb, n8peer);
  f32_to_bf16_kernel<<<n8w / 256, 256, 0, stream>>>(wq, wqb, n8w);
  f32_to_bf16_kernel<<<n8w / 256, 256, 0, stream>>>(wk, wkb, n8w);
  f32_to_bf16_kernel<<<n8w / 256, 256, 0, stream>>>(wv, wvb, n8w);
  f32_to_bf16_kernel<<<n8w / 256, 256, 0, stream>>>(w1, w1b, n8w);
  f32_to_bf16_kernel<<<n8w / 256, 256, 0, stream>>>(w2, w2b, n8w);

  // grid = (M/256 m-groups) x (4 strips); 8 waves/block share one strip
  const int gq  = (BT    / 256) * 4;   // 16 blocks  (1024-row GEMMs)
  const int gkv = (MPEER / 256) * 4;   // 1024 blocks (65536-row GEMMs)

  // Projections: hot K/V GEMMs read pure bf16 (no inner-loop converts)
  gemm_xwt_kernel<0, float , float ><<<gq , 256, 0, stream>>>(target_h, wqb, bq, Q , BT);
  gemm_xwt_kernel<0, __bf16, __bf16><<<gkv, 256, 0, stream>>>(Pb      , wkb, bk, Kp, MPEER);
  gemm_xwt_kernel<0, __bf16, __bf16><<<gkv, 256, 0, stream>>>(Pb      , wvb, bv, Vp, MPEER);

  // Banded attention logits + top-k softmax aggregation
  lag_logits_kernel<<<(B * T * N) / 8, 256, 0, stream>>>(Q, Kp, peer_mask, lg);
  topk_attend_kernel<<<BT, 256, 0, stream>>>(lg, Vp, csy);

  // FFN (WMMA) + residual LayerNorm
  gemm_xwt_kernel<1, float, float><<<gq, 256, 0, stream>>>(csy, w1b, b1, h1, BT);
  gemm_xwt_kernel<0, float, float><<<gq, 256, 0, stream>>>(h1 , w2b, b2, f2, BT);
  resid_ln_kernel<<<BT, 256, 0, stream>>>(csy, f2, gamma, beta, out);
}
